// Block_85753317032260
// MI455X (gfx1250) — matrix operand order repaired and checked by reading, not compiled, not run on hardware
//
#include <hip/hip_runtime.h>
#include <hip/hip_bf16.h>
#include <math.h>

#define D_   1024
#define H_   16
#define HD_  64
#define T_   2048
#define B_   2
#define FF_  4096
#define NROW (B_*T_)   // 4096

typedef __attribute__((ext_vector_type(16))) __bf16         v16bf;
typedef __attribute__((ext_vector_type(8)))  float          v8f;
typedef __attribute__((ext_vector_type(8)))  unsigned short v8us;
typedef __attribute__((ext_vector_type(4)))  unsigned int   v4ui;
typedef __attribute__((ext_vector_type(4)))  int            v4i;
typedef __attribute__((ext_vector_type(8)))  int            v8i;

#if __has_builtin(__builtin_amdgcn_tensor_load_to_lds)
#define HAVE_TDM 1
#else
#define HAVE_TDM 0
#endif

union FragU { v16bf b; v8us u2[2]; unsigned short s[16]; };

__device__ inline unsigned short f2bf(float f) {
  unsigned int u = __float_as_uint(f);
  if ((u & 0x7F800000u) == 0x7F800000u) return (unsigned short)(u >> 16); // inf/nan
  u += 0x7FFFu + ((u >> 16) & 1u);                                       // RNE
  return (unsigned short)(u >> 16);
}

__device__ inline v8f zero8() {
  v8f r;
#pragma unroll
  for (int i = 0; i < 8; ++i) r[i] = 0.0f;
  return r;
}

// A-matrix 16x32 bf16 fragment (ISA 7.12.2): lane<16 -> M=lane, K={h8..h8+7, 16+h8..}
__device__ inline v16bf load_frag_a(const unsigned short* p, int ld, int k0, int lane) {
  const int half = lane >> 4, m = lane & 15;
  const unsigned short* r = p + (size_t)m * ld + k0 + half * 8;
  FragU f;
  f.u2[0] = *(const v8us*)(r);
  f.u2[1] = *(const v8us*)(r + 16);
  return f.b;
}
// B-matrix 32x16 bf16 fragment from an N-major buffer: lane n holds
// K={h8..h8+7, 16+h8..16+h8+7} (same K order as the A fragment)
__device__ inline v16bf load_frag_b(const unsigned short* p, int ld, int k0, int lane) {
  const int n = lane & 15, base = (lane >> 4) << 3;
  const unsigned short* r = p + (size_t)n * ld + k0 + base;
  FragU f;
  f.u2[0] = *(const v8us*)(r);
  f.u2[1] = *(const v8us*)(r + 16);
  return f.b;
}

__device__ inline v8f wmma_bf16(v16bf a, v16bf b, v8f c) {
  return __builtin_amdgcn_wmma_f32_16x16x32_bf16(false, a, false, b, (short)0, c, false, false);
}

// ---------------- Tensor Data Mover: 2D tile (bf16) global -> LDS ----------------
// D# per CDNA5 ISA: group0 = {count/flags, lds_addr,
// global_addr[31:0], global_addr[56:32] | type=2<<30}; group1 packs data_size=2B,
// tensor_dim0=K, tensor_dim1=N, tile_dim0=tk, tile_dim1=tn, tensor_dim0_stride=K.
#if HAVE_TDM
__device__ inline void tdm_load_2d(unsigned lds_off, unsigned long long gaddr,
                                   unsigned K, unsigned N, unsigned tk, unsigned tn) {
  v4ui g0;
  g0[0] = 1u;                                              // count=1 (user descriptor)
  g0[1] = lds_off;                                         // lds_addr (bytes)
  g0[2] = (unsigned)(gaddr & 0xFFFFFFFFu);                 // global_addr[31:0]
  g0[3] = (unsigned)((gaddr >> 32) & 0x01FFFFFFu) | 0x80000000u; // ga[56:32] | type=2
  v8i g1;
  g1[0] = (int)0x00010000;                                 // wg_mask=0, data_size=1 (2B)
  g1[1] = (int)((K & 0xFFFFu) << 16);                      // tensor_dim0[15:0] @ bits 63:48
  g1[2] = (int)(((K >> 16) & 0xFFFFu) | ((N & 0xFFFFu) << 16)); // dim0 hi | dim1 lo
  g1[3] = (int)(((N >> 16) & 0xFFFFu) | ((tk & 0xFFFFu) << 16)); // dim1 hi | tile_dim0
  g1[4] = (int)(tn & 0xFFFFu);                             // tile_dim1 | tile_dim2=0
  g1[5] = (int)K;                                          // tensor_dim0_stride[31:0]
  g1[6] = 0;                                               // stride0 hi | stride1 lo
  g1[7] = 0;
  v4i z4 = {0, 0, 0, 0};
#if defined(__clang_major__) && (__clang_major__ >= 23)
  v8i z8 = {0, 0, 0, 0, 0, 0, 0, 0};
  __builtin_amdgcn_tensor_load_to_lds(g0, g1, z4, z4, z8, 0);
#else
  __builtin_amdgcn_tensor_load_to_lds(g0, g1, z4, z4, 0);
#endif
}
#endif

// ---------------- weight conversion ----------------

__global__ void cvt_qkv_kernel(const float* __restrict__ wq, const float* __restrict__ wk,
                               const float* __restrict__ wv, unsigned short* __restrict__ outT) {
  int idx = blockIdx.x * blockDim.x + threadIdx.x;
  if (idx >= 3 * D_ * D_) return;
  int c = idx % D_;
  int j = idx / D_;           // output column index of C, 0..3D
  int sec = j / D_;
  int jj = j % D_;
  int h = jj / HD_, d = jj % HD_;
  const float* w = (sec == 0) ? wq : (sec == 1 ? wk : wv);
  outT[(size_t)j * D_ + c] = f2bf(w[((size_t)h * D_ + c) * HD_ + d]);
}

__global__ void cvt_t_kernel(const float* __restrict__ in, unsigned short* __restrict__ outT,
                             int K, int N) {
  size_t idx = (size_t)blockIdx.x * blockDim.x + threadIdx.x;
  if (idx >= (size_t)K * N) return;
  int j = (int)(idx / K), c = (int)(idx % K);
  outT[idx] = f2bf(in[(size_t)c * N + j]);
}

// ---------------- layernorm ----------------

__global__ __launch_bounds__(256) void ln_kernel(const float* __restrict__ x,
                                                 const float* __restrict__ g,
                                                 const float* __restrict__ bta,
                                                 unsigned short* __restrict__ out) {
  __shared__ float s1[256], s2[256];
  const int row = blockIdx.x, t = threadIdx.x;
  const float* xr = x + (size_t)row * D_;
  float vals[4], sum = 0.f, sq = 0.f;
#pragma unroll
  for (int j = 0; j < 4; ++j) { float v = xr[t + j * 256]; vals[j] = v; sum += v; sq += v * v; }
  s1[t] = sum; s2[t] = sq; __syncthreads();
  for (int off = 128; off; off >>= 1) {
    if (t < off) { s1[t] += s1[t + off]; s2[t] += s2[t + off]; }
    __syncthreads();
  }
  float mu = s1[0] * (1.0f / D_);
  float var = s2[0] * (1.0f / D_) - mu * mu;
  float rstd = rsqrtf(var + 1e-5f);
#pragma unroll
  for (int j = 0; j < 4; ++j) {
    int i = t + j * 256;
    out[(size_t)row * D_ + i] = f2bf((vals[j] - mu) * rstd * g[i] + bta[i]);
  }
}

// ---------------- WMMA GEMM ----------------
// C[M,N] = A[M,K](bf16) x Bt[N,K](bf16).
// Block: 256 threads = 8 waves, block tile 128(M) x 64(N); wave tile 32x32
// (2 A-frags x 2 B-frags -> 4 WMMAs per K-step). B block-tile (64x32 bf16, 4 KB)
// staged in LDS by the Tensor Data Mover (wave 0 issues, s_wait_tensorcnt, barrier).
// EPI 0: C -> bf16 ; EPI 1: C + bias + res -> fp32 ; EPI 2: gelu(C + bias) -> bf16
template <int EPI>
__global__ __launch_bounds__(256) void gemm_kernel(const unsigned short* __restrict__ A, int lda,
                                                   const unsigned short* __restrict__ Bt, int ldb,
                                                   int M, int N, int K,
                                                   const float* __restrict__ bias,
                                                   const float* __restrict__ res,
                                                   unsigned short* __restrict__ outBf,
                                                   float* __restrict__ outF) {
  __shared__ unsigned short bTile[64][32];
  const int lane = threadIdx.x & 31;
  const int wave = threadIdx.x >> 5;
  const int wm = wave & 3, wn = wave >> 2;
  const int rowTile = blockIdx.y * 128 + wm * 32;
  const int colBlock = blockIdx.x * 64;
  const int colTile = colBlock + wn * 32;
  const unsigned short* aP = A + (size_t)rowTile * lda;
  const unsigned short* bBlk = Bt + (size_t)colBlock * ldb;

  v8f acc00 = zero8(), acc01 = zero8(), acc10 = zero8(), acc11 = zero8();

  for (int k0 = 0; k0 < K; k0 += 32) {
#if HAVE_TDM
    if (wave == 0) {
      tdm_load_2d((unsigned)(size_t)&bTile[0][0],
                  (unsigned long long)(size_t)(bBlk + k0),
                  (unsigned)K, (unsigned)N, 32u, 64u);
      __builtin_amdgcn_s_wait_tensorcnt(0);
    }
#else
    {
      const int t = threadIdx.x;
      const int r = t >> 2, c8 = (t & 3) * 8;
      *(v8us*)&bTile[r][c8] = *(const v8us*)(bBlk + (size_t)r * ldb + k0 + c8);
    }
#endif
    __syncthreads();
    if (k0 + 32 < K) {
      __builtin_prefetch(aP + (size_t)(lane & 15) * lda + k0 + 32, 0, 1);
      __builtin_prefetch(aP + (size_t)((lane & 15) + 16) * lda + k0 + 32, 0, 1);
    }
    v16bf a0 = load_frag_a(aP, lda, k0, lane);
    v16bf a1 = load_frag_a(aP + (size_t)16 * lda, lda, k0, lane);
    v16bf b0 = load_frag_b(&bTile[wn * 32][0], 32, 0, lane);
    v16bf b1 = load_frag_b(&bTile[wn * 32 + 16][0], 32, 0, lane);
    acc00 = wmma_bf16(a0, b0, acc00);
    acc01 = wmma_bf16(a0, b1, acc01);
    acc10 = wmma_bf16(a1, b0, acc10);
    acc11 = wmma_bf16(a1, b1, acc11);
    __syncthreads();
  }

  const int half = lane >> 4, n = lane & 15;
#pragma unroll
  for (int j = 0; j < 2; ++j) {
    const int col = colTile + j * 16 + n;
    const float bv = (EPI != 0) ? bias[col] : 0.0f;
    const v8f accT = j ? acc01 : acc00;
    const v8f accB = j ? acc11 : acc10;
#pragma unroll
    for (int r = 0; r < 8; ++r) {
      const int rowT = rowTile + r + half * 8;
      const int rowB = rowT + 16;
      const float vT = accT[r], vB = accB[r];
      if (EPI == 0) {
        outBf[(size_t)rowT * N + col] = f2bf(vT);
        outBf[(size_t)rowB * N + col] = f2bf(vB);
      } else if (EPI == 1) {
        outF[(size_t)rowT * N + col] = vT + bv + res[(size_t)rowT * N + col];
        outF[(size_t)rowB * N + col] = vB + bv + res[(size_t)rowB * N + col];
      } else {
        const float zT = vT + bv, zB = vB + bv;
        outBf[(size_t)rowT * N + col] = f2bf(0.5f * zT * (1.0f + erff(zT * 0.70710678118f)));
        outBf[(size_t)rowB * N + col] = f2bf(0.5f * zB * (1.0f + erff(zB * 0.70710678118f)));
      }
    }
  }
}

// ---------------- flash attention ----------------
// one wave per (b, h, 16-query tile); online softmax; causal; P staged via LDS to
// convert C/D layout -> A layout; V tile staged in LDS. All short loops unrolled
// so per-row stats live in static registers (no v_movrels).
__global__ __launch_bounds__(32) void attn_kernel(const unsigned short* __restrict__ qkv,
                                                  unsigned short* __restrict__ attnout) {
  __shared__ unsigned short ptile[16 * 32];
  __shared__ unsigned short vtile[32 * 64];
  const int lane = threadIdx.x & 31;
  const int t0 = blockIdx.x * 16;
  const int h = blockIdx.y;
  const int b = blockIdx.z;
  const int half = lane >> 4, n = lane & 15;
  const float scale = 0.125f;  // HD^-0.5

  const unsigned short* qbase = qkv + ((size_t)(b * T_ + t0)) * (3 * D_) + h * HD_;
  const v16bf qf0 = load_frag_a(qbase, 3 * D_, 0, lane);
  const v16bf qf1 = load_frag_a(qbase, 3 * D_, 32, lane);

  float mrow[8], lrow[8];
  v8f o[4];
#pragma unroll
  for (int r = 0; r < 8; ++r) { mrow[r] = -INFINITY; lrow[r] = 0.0f; }
#pragma unroll
  for (int nt = 0; nt < 4; ++nt) o[nt] = zero8();

  for (int k0 = 0; k0 <= t0 + 15; k0 += 32) {
    const unsigned short* kb = qkv + ((size_t)(b * T_ + k0)) * (3 * D_) + D_ + h * HD_;
    v8f s0 = zero8(), s1 = zero8();
    {
      v16bf b00 = load_frag_b(kb, 3 * D_, 0, lane);
      v16bf b01 = load_frag_b(kb, 3 * D_, 32, lane);
      v16bf b10 = load_frag_b(kb + (size_t)16 * (3 * D_), 3 * D_, 0, lane);
      v16bf b11 = load_frag_b(kb + (size_t)16 * (3 * D_), 3 * D_, 32, lane);
      s0 = wmma_bf16(qf0, b00, s0);
      s0 = wmma_bf16(qf1, b01, s0);
      s1 = wmma_bf16(qf0, b10, s1);
      s1 = wmma_bf16(qf1, b11, s1);
    }

    // stage V tile [32 keys x 64 cols] in LDS (lane owns key row k0+lane)
    const unsigned short* vsrc =
        qkv + ((size_t)(b * T_ + k0 + lane)) * (3 * D_) + 2 * D_ + h * HD_;
#pragma unroll
    for (int i = 0; i < 8; ++i)
      *(v8us*)&vtile[lane * 64 + i * 8] = *(const v8us*)(vsrc + i * 8);

    // causal mask + scale + online softmax; write P (bf16) to LDS
#pragma unroll
    for (int r = 0; r < 8; ++r) {
      const int row = t0 + r + half * 8;
      float a0 = s0[r] * scale; if (k0 + n > row)      a0 = -INFINITY;
      float a1 = s1[r] * scale; if (k0 + 16 + n > row) a1 = -INFINITY;
      float mx = fmaxf(a0, a1);
#pragma unroll
      for (int off = 8; off; off >>= 1) mx = fmaxf(mx, __shfl_xor(mx, off, 16));
      const float nm = fmaxf(mrow[r], mx);
      const float alpha = __expf(mrow[r] - nm);
      const float p0 = __expf(a0 - nm);
      const float p1 = __expf(a1 - nm);
      float rs = p0 + p1;
#pragma unroll
      for (int off = 8; off; off >>= 1) rs += __shfl_xor(rs, off, 16);
      mrow[r] = nm;
      lrow[r] = lrow[r] * alpha + rs;
      o[0][r] *= alpha; o[1][r] *= alpha; o[2][r] *= alpha; o[3][r] *= alpha;
      ptile[(r + half * 8) * 32 + n]      = f2bf(p0);
      ptile[(r + half * 8) * 32 + 16 + n] = f2bf(p1);
    }
    __syncthreads();

    // P as A-fragment (16x32), V columns as B-fragments, accumulate O
    FragU pf;
    pf.u2[0] = *(const v8us*)&ptile[(lane & 15) * 32 + half * 8];
    pf.u2[1] = *(const v8us*)&ptile[(lane & 15) * 32 + 16 + half * 8];
#pragma unroll
    for (int nt = 0; nt < 4; ++nt) {
      FragU vf;
#pragma unroll
      for (int e = 0; e < 16; ++e)
        vf.s[e] = vtile[(half * 8 + e + (e >= 8 ? 8 : 0)) * 64 + nt * 16 + n];
      o[nt] = wmma_bf16(pf.b, vf.b, o[nt]);
    }
    __syncthreads();
  }

#pragma unroll
  for (int nt = 0; nt < 4; ++nt)
#pragma unroll
    for (int r = 0; r < 8; ++r) {
      const int row = t0 + r + half * 8;
      const int col = h * HD_ + nt * 16 + n;
      attnout[((size_t)(b * T_ + row)) * D_ + col] = f2bf(o[nt][r] / lrow[r]);
    }
}

// ---------------- launch ----------------

extern "C" void kernel_launch(void* const* d_in, const int* in_sizes, int n_in,
                              void* d_out, int out_size, void* d_ws, size_t ws_size,
                              hipStream_t stream) {
  (void)in_sizes; (void)n_in; (void)out_size; (void)ws_size;
  const float* x      = (const float*)d_in[0];
  const float* wq     = (const float*)d_in[1];
  const float* wk     = (const float*)d_in[2];
  const float* wv     = (const float*)d_in[3];
  const float* w_proj = (const float*)d_in[4];
  const float* b_proj = (const float*)d_in[5];
  const float* ln1_g  = (const float*)d_in[6];
  const float* ln1_b  = (const float*)d_in[7];
  const float* ln2_g  = (const float*)d_in[8];
  const float* ln2_b  = (const float*)d_in[9];
  const float* w1     = (const float*)d_in[10];
  const float* b1     = (const float*)d_in[11];
  const float* w2     = (const float*)d_in[12];
  const float* b2     = (const float*)d_in[13];
  float* out = (float*)d_out;

  char* ws = (char*)d_ws;
  size_t off = 0;
  auto alloc = [&](size_t bytes) {
    void* p = ws + off;
    off += (bytes + 255) & ~(size_t)255;
    return p;
  };
  unsigned short* hbf    = (unsigned short*)alloc((size_t)NROW * D_ * 2);
  unsigned short* wqkvT  = (unsigned short*)alloc((size_t)3 * D_ * D_ * 2);
  unsigned short* qkv    = (unsigned short*)alloc((size_t)NROW * 3 * D_ * 2);
  unsigned short* attnO  = (unsigned short*)alloc((size_t)NROW * D_ * 2);
  unsigned short* wprojT = (unsigned short*)alloc((size_t)D_ * D_ * 2);
  float*          x1     = (float*)alloc((size_t)NROW * D_ * 4);
  unsigned short* h2bf   = (unsigned short*)alloc((size_t)NROW * D_ * 2);
  unsigned short* w1T    = (unsigned short*)alloc((size_t)D_ * FF_ * 2);
  unsigned short* ffbf   = (unsigned short*)alloc((size_t)NROW * FF_ * 2);
  unsigned short* w2T    = (unsigned short*)alloc((size_t)FF_ * D_ * 2);

  // weight conversions (deterministic each call; graph-capture safe)
  { int tot = 3 * D_ * D_;
    cvt_qkv_kernel<<<(tot + 255) / 256, 256, 0, stream>>>(wq, wk, wv, wqkvT); }
  { size_t tot = (size_t)D_ * D_;
    cvt_t_kernel<<<(unsigned)((tot + 255) / 256), 256, 0, stream>>>(w_proj, wprojT, D_, D_); }
  { size_t tot = (size_t)D_ * FF_;
    cvt_t_kernel<<<(unsigned)((tot + 255) / 256), 256, 0, stream>>>(w1, w1T, D_, FF_); }
  { size_t tot = (size_t)FF_ * D_;
    cvt_t_kernel<<<(unsigned)((tot + 255) / 256), 256, 0, stream>>>(w2, w2T, FF_, D_); }

  // LN1
  ln_kernel<<<NROW, 256, 0, stream>>>(x, ln1_g, ln1_b, hbf);

  // QKV projection: [4096,1024] x [1024,3072]
  gemm_kernel<0><<<dim3(3 * D_ / 64, NROW / 128), 256, 0, stream>>>(
      hbf, D_, wqkvT, D_, NROW, 3 * D_, D_, nullptr, nullptr, qkv, nullptr);

  // flash attention
  attn_kernel<<<dim3(T_ / 16, H_, B_), 32, 0, stream>>>(qkv, attnO);

  // output projection + bias + residual -> x1 (fp32)
  gemm_kernel<1><<<dim3(D_ / 64, NROW / 128), 256, 0, stream>>>(
      attnO, D_, wprojT, D_, NROW, D_, D_, b_proj, x, nullptr, x1);

  // LN2
  ln_kernel<<<NROW, 256, 0, stream>>>(x1, ln2_g, ln2_b, h2bf);

  // FF1 + bias + GELU -> bf16
  gemm_kernel<2><<<dim3(FF_ / 64, NROW / 128), 256, 0, stream>>>(
      h2bf, D_, w1T, D_, NROW, FF_, D_, b1, nullptr, ffbf, nullptr);

  // FF2 + bias + residual -> d_out (fp32)
  gemm_kernel<1><<<dim3(D_ / 64, NROW / 128), 256, 0, stream>>>(
      ffbf, FF_, w2T, FF_, NROW, D_, FF_, b2, x1, nullptr, out);
}
